// WD_LSTM_63711544869525
// MI455X (gfx1250) — matrix operand order repaired and checked by reading, not compiled, not run on hardware
//
#include <hip/hip_runtime.h>
#include <hip/hip_bf16.h>
#include <math.h>

// Problem dims (fixed by the reference)
#define T_STEPS 2048
#define D_IN    512
#define U_HID   1024
#define G4      4096      // 4*U
#define V_OUT   50000
#define NWG_SCAN 16       // persistent workgroups for the sequential scan

typedef __attribute__((ext_vector_type(16))) __bf16 v16bf;
typedef __attribute__((ext_vector_type(8)))  float  v8f;
typedef __attribute__((ext_vector_type(2)))  float  v2f;

union BF16Frag { unsigned u[8]; v16bf v; };

// ---------------------------------------------------------------------------
// Prep kernels: gather+cast embeddings, cast w_ih0 to bf16, merge upper weights
// ---------------------------------------------------------------------------
__global__ __launch_bounds__(256)
void k_gather_cast(const int* __restrict__ ids, const float* __restrict__ X,
                   __bf16* __restrict__ Xg) {
    const int t = blockIdx.x;
    const int node = ids[t];
    const float* src = X + (size_t)node * D_IN;
    __bf16* dst = Xg + (size_t)t * D_IN;
    for (int d = threadIdx.x; d < D_IN; d += blockDim.x)
        dst[d] = (__bf16)src[d];
}

__global__ __launch_bounds__(256)
void k_cast_bf16(const float* __restrict__ src, __bf16* __restrict__ dst, int n) {
    int i = blockIdx.x * blockDim.x + threadIdx.x;
    if (i < n) dst[i] = (__bf16)src[i];
}

__global__ __launch_bounds__(256)
void k_merge_add(const float* __restrict__ a, const float* __restrict__ b,
                 float* __restrict__ o, int n) {
    int i = blockIdx.x * blockDim.x + threadIdx.x;
    if (i < n) o[i] = a[i] + b[i];
}

// ---------------------------------------------------------------------------
// Kernel A: input projection GEMM via bf16 WMMA
//   P[t][g] = sum_k Xg[t][k] * Wb[g][k]  + b_ih0[g] + b_hh0[g]
//   Wave computes 16(t) x 64(g); 8 waves per WG -> 128 x 64 tile.
// ---------------------------------------------------------------------------
__global__ __launch_bounds__(256)
void k_inproj_wmma(const __bf16* __restrict__ Xg, const __bf16* __restrict__ Wb,
                   const float* __restrict__ b_ih0, const float* __restrict__ b_hh0,
                   float* __restrict__ P) {
    const int lane = threadIdx.x & 31;
    const int wave = threadIdx.x >> 5;
    const int half = lane >> 4;      // 0: lanes 0-15, 1: lanes 16-31
    const int nl   = lane & 15;
    const int t0   = blockIdx.y * 128 + wave * 16;
    const int g0   = blockIdx.x * 64;

    v8f c[4] = {{0}, {0}, {0}, {0}};
    const __bf16* arow = Xg + (size_t)(t0 + nl) * D_IN;

    for (int k0 = 0; k0 < D_IN; k0 += 32) {
        // A fragment: 16x32 bf16 (M x K). Lane nl = row M; per-VGPR K pairs:
        //   VGPR r: K = k0 + half*16 + 2r
        BF16Frag a;
#pragma unroll
        for (int r = 0; r < 8; ++r) {
            int k = k0 + half * 16 + r * 2;
            a.u[r] = *(const unsigned*)(arow + k);
        }
#pragma unroll
        for (int s = 0; s < 4; ++s) {
            // B fragment: 32x16 bf16 (K x N). Lane nl = col N;
            //   lanes 0-15 hold K=k0..k0+15, lanes 16-31 hold K=k0+16..k0+31.
            const int n = g0 + s * 16 + nl;
            const __bf16* brow = Wb + (size_t)n * D_IN + k0 + half * 16;
            BF16Frag b;
#pragma unroll
            for (int r = 0; r < 8; ++r)
                b.u[r] = *(const unsigned*)(brow + r * 2);
            c[s] = __builtin_amdgcn_wmma_f32_16x16x32_bf16(
                false, a.v, false, b.v, (short)0, c[s], false, false);
        }
    }

    // C/D layout: VGPR j -> M = j + 8*half, N = nl
#pragma unroll
    for (int s = 0; s < 4; ++s) {
        const int col = g0 + s * 16 + nl;
        const float bias = b_ih0[col] + b_hh0[col];
#pragma unroll
        for (int j = 0; j < 8; ++j) {
            const int row = t0 + j + 8 * half;
            P[(size_t)row * G4 + col] = c[s][j] + bias;
        }
    }
}

// ---------------------------------------------------------------------------
// Sequential LSTM scan: 16 persistent WGs, grid-wide atomic barrier.
// Each thread owns one of the 4096 gate rows; cell updates are redundantly
// recomputed per-WG into private LDS state (h0, c0, c) -> 3 barriers/step.
// Gate scratch is triple-buffered (one buffer per layer) so no extra barrier
// is needed between the update-read and the next layer's write.
// ---------------------------------------------------------------------------
__device__ __forceinline__ float sigmoidf_(float x) {
    return 1.0f / (1.0f + __expf(-x));
}

__device__ __forceinline__ void grid_sync(unsigned* cnt, unsigned idx) {
    __syncthreads();
    if (threadIdx.x == 0) {
        __threadfence();
        __hip_atomic_fetch_add(cnt, 1u, __ATOMIC_RELEASE, __HIP_MEMORY_SCOPE_AGENT);
        const unsigned goal = idx * NWG_SCAN;
        while (__hip_atomic_load(cnt, __ATOMIC_ACQUIRE, __HIP_MEMORY_SCOPE_AGENT) < goal)
            __builtin_amdgcn_s_sleep(2);
        __threadfence();
    }
    __syncthreads();
}

__global__ __launch_bounds__(256)
void k_scan(const float* __restrict__ P,     // [T][4096] incl. layer-0 biases
            const float* __restrict__ whh0,  // [4096][1024] fp32
            const float* __restrict__ Wm,    // [2][4096][1024] merged upper
            const float* __restrict__ bm,    // [2][4096] merged upper biases
            float* __restrict__ hs,          // [T][1024] output hidden
            float* __restrict__ gbuf,        // [3][4096] gate scratch
            unsigned* __restrict__ cnt) {
    __shared__ float xsh[U_HID];   // matvec input for upper layers
    __shared__ float h0k[U_HID];   // persistent layer-0 hidden (per-WG copy)
    __shared__ float c0k[U_HID];   // persistent layer-0 cell
    __shared__ float cck[U_HID];   // within-step cell carried up the stack

    const int tid = threadIdx.x;
    const int j   = blockIdx.x * 256 + tid;     // gate row in [0,4096)

    for (int u = tid; u < U_HID; u += 256) { h0k[u] = 0.f; c0k[u] = 0.f; }
    __syncthreads();

    unsigned bidx = 0;
    for (int t = 0; t < T_STEPS; ++t) {
        // ----- layer 0 gates: P[t] + whh0 @ h0 -----
        {
            const float4* wr = (const float4*)(whh0 + (size_t)j * U_HID);
            const float4* xr = (const float4*)h0k;
            float acc = P[(size_t)t * G4 + j];
#pragma unroll 8
            for (int q = 0; q < U_HID / 4; ++q) {
                float4 w4 = wr[q], x4 = xr[q];
                acc += w4.x * x4.x + w4.y * x4.y + w4.z * x4.z + w4.w * x4.w;
            }
            gbuf[j] = acc;
        }
        grid_sync(cnt, ++bidx);
        // ----- layer 0 cell update (redundant in every WG) -----
        for (int u = tid; u < U_HID; u += 256) {
            float gi = sigmoidf_(gbuf[u]);
            float gf = sigmoidf_(gbuf[u + U_HID]);
            float gg = tanhf    (gbuf[u + 2 * U_HID]);
            float go = sigmoidf_(gbuf[u + 3 * U_HID]);
            float cn = gf * c0k[u] + gi * gg;
            float hn = go * tanhf(cn);
            c0k[u] = cn; h0k[u] = hn; cck[u] = cn; xsh[u] = hn;
        }
        __syncthreads();
        // ----- upper layers: gates = (w_ih+w_hh) @ h + (b_ih+b_hh) -----
#pragma unroll
        for (int l = 0; l < 2; ++l) {
            {
                const float4* wr =
                    (const float4*)(Wm + ((size_t)l * G4 + j) * U_HID);
                const float4* xr = (const float4*)xsh;
                float acc = bm[l * G4 + j];
#pragma unroll 8
                for (int q = 0; q < U_HID / 4; ++q) {
                    float4 w4 = wr[q], x4 = xr[q];
                    acc += w4.x * x4.x + w4.y * x4.y + w4.z * x4.z + w4.w * x4.w;
                }
                gbuf[(l + 1) * G4 + j] = acc;
            }
            grid_sync(cnt, ++bidx);
            for (int u = tid; u < U_HID; u += 256) {
                const float* gb = gbuf + (l + 1) * G4;
                float gi = sigmoidf_(gb[u]);
                float gf = sigmoidf_(gb[u + U_HID]);
                float gg = tanhf    (gb[u + 2 * U_HID]);
                float go = sigmoidf_(gb[u + 3 * U_HID]);
                float cn = gf * cck[u] + gi * gg;
                float hn = go * tanhf(cn);
                cck[u] = cn; xsh[u] = hn;
                if (l == 1 && blockIdx.x == 0)
                    hs[(size_t)t * U_HID + u] = hn;
            }
            __syncthreads();
        }
    }
}

// ---------------------------------------------------------------------------
// Kernel C: output GEMM via f32 WMMA (V_WMMA_F32_16X16X4_F32), reading w_g
// fp32 directly (no 100MB bf16 copy).  logits[t][v] = hs[t]·w_g[v] + b_g[v].
// Wave: 16(t) x 64(v); 8 waves/WG -> 128 x 64.
// ---------------------------------------------------------------------------
__global__ __launch_bounds__(256)
void k_outproj_wmma(const float* __restrict__ hsv, const float* __restrict__ wg,
                    const float* __restrict__ bg, float* __restrict__ out) {
    const int lane = threadIdx.x & 31;
    const int wave = threadIdx.x >> 5;
    const int half = lane >> 4;
    const int nl   = lane & 15;
    const int t0   = blockIdx.y * 128 + wave * 16;
    const int v0   = blockIdx.x * 64;

    v8f c[4] = {{0}, {0}, {0}, {0}};
    // A 16x4 f32: lane nl = row M; lanes 0-15 hold K={k,k+1}, lanes 16-31 {k+2,k+3}
    const float* arow = hsv + (size_t)(t0 + nl) * U_HID + 2 * half;

    for (int k = 0; k < U_HID; k += 4) {
        v2f a = *(const v2f*)(arow + k);
#pragma unroll
        for (int s = 0; s < 4; ++s) {
            const int n = v0 + s * 16 + nl;
            v2f b = {0.f, 0.f};
            if (n < V_OUT)   // B 4x16 f32: lane nl = col N, same K split as A
                b = *(const v2f*)(wg + (size_t)n * U_HID + k + 2 * half);
            c[s] = __builtin_amdgcn_wmma_f32_16x16x4_f32(
                false, a, false, b, (short)0, c[s], false, false);
        }
    }

#pragma unroll
    for (int s = 0; s < 4; ++s) {
        const int col = v0 + s * 16 + nl;
        if (col < V_OUT) {
            const float bias = bg[col];
#pragma unroll
            for (int j = 0; j < 8; ++j) {
                const int row = t0 + j + 8 * half;
                out[(size_t)row * V_OUT + col] = c[s][j] + bias;
            }
        }
    }
}

// ---------------------------------------------------------------------------
// Log-softmax in place: one WG per row of [2048][50000].
// ---------------------------------------------------------------------------
__global__ __launch_bounds__(256)
void k_logsoftmax(float* __restrict__ out) {
    __shared__ float red[256];
    const int t = blockIdx.x;
    float* row = out + (size_t)t * V_OUT;

    float m = -INFINITY;
    for (int v = threadIdx.x; v < V_OUT; v += 256) m = fmaxf(m, row[v]);
    red[threadIdx.x] = m;
    __syncthreads();
    for (int s = 128; s > 0; s >>= 1) {
        if (threadIdx.x < s)
            red[threadIdx.x] = fmaxf(red[threadIdx.x], red[threadIdx.x + s]);
        __syncthreads();
    }
    m = red[0];
    __syncthreads();

    float sum = 0.f;
    for (int v = threadIdx.x; v < V_OUT; v += 256) sum += __expf(row[v] - m);
    red[threadIdx.x] = sum;
    __syncthreads();
    for (int s = 128; s > 0; s >>= 1) {
        if (threadIdx.x < s) red[threadIdx.x] += red[threadIdx.x + s];
        __syncthreads();
    }
    const float lse = m + __logf(red[0]);
    __syncthreads();

    for (int v = threadIdx.x; v < V_OUT; v += 256) row[v] -= lse;
}

// ---------------------------------------------------------------------------
extern "C" void kernel_launch(void* const* d_in, const int* in_sizes, int n_in,
                              void* d_out, int out_size, void* d_ws, size_t ws_size,
                              hipStream_t stream) {
    (void)in_sizes; (void)n_in; (void)out_size; (void)ws_size;
    const int*   ids     = (const int*)  d_in[0];
    const float* X       = (const float*)d_in[1];
    const float* w_ih0   = (const float*)d_in[2];
    const float* w_hh0   = (const float*)d_in[3];
    const float* b_ih0   = (const float*)d_in[4];
    const float* b_hh0   = (const float*)d_in[5];
    const float* w_ih_up = (const float*)d_in[6];
    const float* w_hh_up = (const float*)d_in[7];
    const float* b_ih_up = (const float*)d_in[8];
    const float* b_hh_up = (const float*)d_in[9];
    const float* w_g     = (const float*)d_in[10];
    const float* b_g     = (const float*)d_in[11];
    float* out = (float*)d_out;

    // Workspace carve-out (~82 MB total), 256B aligned regions.
    char* ws = (char*)d_ws;
    size_t off = 0;
    auto carve = [&](size_t bytes) -> void* {
        void* p = ws + off;
        off = (off + bytes + 255) & ~(size_t)255;
        return p;
    };
    unsigned* cnt  = (unsigned*)carve(256);
    __bf16*   Xg   = (__bf16*) carve((size_t)T_STEPS * D_IN * sizeof(__bf16));
    __bf16*   Wb   = (__bf16*) carve((size_t)G4 * D_IN * sizeof(__bf16));
    float*    P    = (float*)  carve((size_t)T_STEPS * G4 * sizeof(float));
    float*    Wm   = (float*)  carve((size_t)2 * G4 * U_HID * sizeof(float));
    float*    bm   = (float*)  carve((size_t)2 * G4 * sizeof(float));
    float*    hsv  = (float*)  carve((size_t)T_STEPS * U_HID * sizeof(float));
    float*    gbuf = (float*)  carve((size_t)3 * G4 * sizeof(float));

    (void)hipMemsetAsync(cnt, 0, 256, stream);   // reset grid barrier each call

    k_gather_cast<<<T_STEPS, 256, 0, stream>>>(ids, X, Xg);
    k_cast_bf16<<<(G4 * D_IN + 255) / 256, 256, 0, stream>>>(w_ih0, Wb, G4 * D_IN);
    k_merge_add<<<(2 * G4 * U_HID + 255) / 256, 256, 0, stream>>>(
        w_ih_up, w_hh_up, Wm, 2 * G4 * U_HID);
    k_merge_add<<<(2 * G4 + 255) / 256, 256, 0, stream>>>(
        b_ih_up, b_hh_up, bm, 2 * G4);

    dim3 gA(G4 / 64, T_STEPS / 128);
    k_inproj_wmma<<<gA, 256, 0, stream>>>(Xg, Wb, b_ih0, b_hh0, P);

    k_scan<<<NWG_SCAN, 256, 0, stream>>>(P, w_hh0, Wm, bm, hsv, gbuf, cnt);

    dim3 gC((V_OUT + 63) / 64, T_STEPS / 128);
    k_outproj_wmma<<<gC, 256, 0, stream>>>(hsv, w_g, b_g, out);

    k_logsoftmax<<<T_STEPS, 256, 0, stream>>>(out);
}
